// CCAD_90520730730768
// MI455X (gfx1250) — matrix operand order repaired and checked by reading, not compiled, not run on hardware
//
#include <hip/hip_runtime.h>
#include <hip/hip_bf16.h>
#include <math.h>

typedef __attribute__((ext_vector_type(16))) __bf16 v16bf;
typedef __attribute__((ext_vector_type(8)))  float  v8f;

#define HDIM 256
#define LDIM 1024
#define CIN  1024
#define BATCH 64
#define TSTEPS 64
#define NCLS 111
#define XDIM 623   // 256 emb + 256 ctx + 111 count

// ---- fp32 -> bf16 (round to nearest even), packed pair ----
__device__ __forceinline__ unsigned int f2bf(float f) {
  unsigned int u = __float_as_uint(f);
  unsigned int r = u + 0x7FFFu + ((u >> 16) & 1u);
  return r >> 16;
}
__device__ __forceinline__ unsigned int packbf(float a, float b) {
  return f2bf(a) | (f2bf(b) << 16);
}

// ============================================================================
// WMMA GEMM:  out[b, l, h] = sum_c A[b, c, l] * W[h, c]  + bias[l, h]
//   A: fp32 (BATCH, CIN, LDIM)   W: fp32 (256 rows used, CIN) row-major
//   out: fp32 (BATCH, LDIM, 256); bias row stride = bias_stride (cols 0..255)
// 256-thread block = 8 waves covering the full N=256; each wave computes a
// 16(l) x 32(h) register-blocked tile (2 accumulators, 2 v_wmma per K-step).
// A tile staged once per block per K-step; bf16 pairs packed into u32 LDS.
// ============================================================================
__global__ void __launch_bounds__(256)
gemm_bf16_wmma(const float* __restrict__ A, const float* __restrict__ W,
               const float* __restrict__ bias, int bias_stride,
               float* __restrict__ out) {
  // kpair index: kp holds K = {2kp, 2kp+1} packed in one u32
  __shared__ unsigned int ldsA[16 * 16];       // [m][kp]      (16 l x 32 k)
  __shared__ unsigned int ldsB[8][32 * 16];    // per-wave [n][kp] (32 h x 32 k)
  const int tid  = threadIdx.x;
  const int lane = tid & 31;
  const int wv   = tid >> 5;
  const int h0   = wv * 32;
  const int l0   = blockIdx.x * 16;
  const int b    = blockIdx.y;
  const float* Ab = A + (size_t)b * CIN * LDIM;

  v8f acc0 = {0.f, 0.f, 0.f, 0.f, 0.f, 0.f, 0.f, 0.f};
  v8f acc1 = {0.f, 0.f, 0.f, 0.f, 0.f, 0.f, 0.f, 0.f};

  for (int c0 = 0; c0 < CIN; c0 += 32) {
    // ---- stage A tile: 256 packed pairs, one per thread ----
    {
      int m = tid & 15, kp = tid >> 4;
      float f0 = Ab[(size_t)(c0 + 2 * kp) * LDIM + l0 + m];
      float f1 = Ab[(size_t)(c0 + 2 * kp + 1) * LDIM + l0 + m];
      ldsA[m * 16 + kp] = packbf(f0, f1);
    }
    // ---- stage B tile per wave: 32n x 16kp = 512 pairs, 16 per lane ----
    for (int e = lane; e < 512; e += 32) {
      int n = e >> 4, kp = e & 15;
      const float2 f = *reinterpret_cast<const float2*>(
          &W[(size_t)(h0 + n) * CIN + c0 + 2 * kp]);
      ldsB[wv][n * 16 + kp] = packbf(f.x, f.y);
    }
    __syncthreads();

    // ---- A fragment: lanes 0-15 row m K0-15; lanes 16-31 K16-31
    v16bf afrag, bfrag0, bfrag1;
    {
      int m  = lane & 15;
      int kb = (lane < 16) ? 0 : 8;   // kpair base: K{0-15}->kp0-7, K{16-31}->kp8-15
      uint4 lo = *reinterpret_cast<const uint4*>(&ldsA[m * 16 + kb]);
      uint4 hi = *reinterpret_cast<const uint4*>(&ldsA[m * 16 + kb + 4]);
      uint4* p = reinterpret_cast<uint4*>(&afrag);
      p[0] = lo; p[1] = hi;
    }
    // ---- B fragments: lanes 0-15 col n K0-15 (kp0-7); lanes 16-31 K16-31
    {
      int n  = lane & 15;
      int kb = (lane < 16) ? 0 : 8;
      uint4 lo = *reinterpret_cast<const uint4*>(&ldsB[wv][n * 16 + kb]);
      uint4 hi = *reinterpret_cast<const uint4*>(&ldsB[wv][n * 16 + kb + 4]);
      uint4* p = reinterpret_cast<uint4*>(&bfrag0);
      p[0] = lo; p[1] = hi;
      uint4 lo1 = *reinterpret_cast<const uint4*>(&ldsB[wv][(n + 16) * 16 + kb]);
      uint4 hi1 = *reinterpret_cast<const uint4*>(&ldsB[wv][(n + 16) * 16 + kb + 4]);
      uint4* q = reinterpret_cast<uint4*>(&bfrag1);
      q[0] = lo1; q[1] = hi1;
    }
    acc0 = __builtin_amdgcn_wmma_f32_16x16x32_bf16(false, afrag, false, bfrag0,
                                                   (short)0, acc0, false, false);
    acc1 = __builtin_amdgcn_wmma_f32_16x16x32_bf16(false, afrag, false, bfrag1,
                                                   (short)0, acc1, false, false);
    __syncthreads();
  }

  // D layout: VGPR r -> row M = r (+8 for lanes 16-31), col N = lane&15
  int nn    = lane & 15;
  int mbase = (lane < 16) ? 0 : 8;
#pragma unroll
  for (int r = 0; r < 8; ++r) {
    int l = l0 + mbase + r;
    size_t orow = (size_t)b * LDIM * HDIM + (size_t)l * HDIM;
    size_t brow = (size_t)l * bias_stride;
    out[orow + h0 + nn]      = acc0[r] + bias[brow + h0 + nn];
    out[orow + h0 + 16 + nn] = acc1[r] + bias[brow + h0 + 16 + nn];
  }
}

// ============================================================================
// Precompute kernels
// ============================================================================
// peB[l, d] = positional_encoding(l, d) + conv_b[d]   (L x 512)
__global__ void peB_kernel(const float* __restrict__ conv_b, float* __restrict__ peB) {
  int l = blockIdx.x, i = threadIdx.x;  // i: 0..255
  float freq = expf(-(float)(2 * i) * (logf(10000.f) / 512.f));
  float ang  = (float)l * freq;
  peB[l * 512 + 2 * i]     = sinf(ang) + conv_b[2 * i];
  peB[l * 512 + 2 * i + 1] = cosf(ang) + conv_b[2 * i + 1];
}

// Wca[h, c] = sum_d attn_w[h, d] * conv_w[d, c]   (256 x 1024)
__global__ void wca_kernel(const float* __restrict__ attn_w,
                           const float* __restrict__ conv_w, float* __restrict__ Wca) {
  int h = blockIdx.x;
  int c = blockIdx.y * 256 + threadIdx.x;
  float s = 0.f;
  for (int d = 0; d < 512; ++d) s += attn_w[h * 512 + d] * conv_w[(size_t)d * 1024 + c];
  Wca[(size_t)h * 1024 + c] = s;
}

// peW[l, h] = sum_d peB[l, d] * attn_w[h, d] + attn_b[h] + cov_b[h]
__global__ void peW_kernel(const float* __restrict__ peB, const float* __restrict__ attn_w,
                           const float* __restrict__ attn_b, const float* __restrict__ cov_b,
                           float* __restrict__ peW) {
  __shared__ float sp[512];
  int l = blockIdx.x, h = threadIdx.x;
  sp[h] = peB[l * 512 + h];
  sp[256 + h] = peB[l * 512 + 256 + h];
  __syncthreads();
  float s = attn_b[h] + cov_b[h];
  const float* aw = attn_w + (size_t)h * 512;
  for (int d = 0; d < 512; ++d) s += sp[d] * aw[d];
  peW[l * 256 + h] = s;
}

__global__ void init_kernel(float* __restrict__ hstate, float* __restrict__ cov) {
  int i = blockIdx.x * 256 + threadIdx.x;
  if (i < BATCH * HDIM) hstate[i] = 0.f;
  if (i < BATCH * LDIM) cov[i] = 0.f;
}

// ============================================================================
// Per-step attention scores: one wave per (b, l)
// scores[b,l] = vb + sum_h v[h] * tanh(Wf'[b,l,h] + h[b,h] + cov[b,l]*cov_w[h])
// ============================================================================
__global__ void __launch_bounds__(256)
scores_kernel(const float* __restrict__ Wf, const float* __restrict__ hstate,
              const float* __restrict__ cov, const float* __restrict__ cov_w,
              const float* __restrict__ attn_v, const float* __restrict__ attn_vb,
              float* __restrict__ scores) {
  int wv = threadIdx.x >> 5, lane = threadIdx.x & 31;
  int p = blockIdx.x * 8 + wv;              // p = b*L + l
  int b = p >> 10;
  const float* wf = Wf + (size_t)p * HDIM;
  const float* hs = hstate + b * HDIM;
  float cv   = cov[p];
  float part = 0.f;
#pragma unroll
  for (int g = 0; g < 2; ++g) {
    int o = g * 128 + lane * 4;
    float4 w  = *reinterpret_cast<const float4*>(wf + o);
    float4 h4 = *reinterpret_cast<const float4*>(hs + o);
    float4 cw = *reinterpret_cast<const float4*>(cov_w + o);
    float4 av = *reinterpret_cast<const float4*>(attn_v + o);
    part += av.x * tanhf(w.x + h4.x + cv * cw.x);
    part += av.y * tanhf(w.y + h4.y + cv * cw.y);
    part += av.z * tanhf(w.z + h4.z + cv * cw.z);
    part += av.w * tanhf(w.w + h4.w + cv * cw.w);
  }
  for (int off = 16; off > 0; off >>= 1) part += __shfl_xor(part, off, 32);
  if (lane == 0) scores[p] = part + attn_vb[0];
}

// ============================================================================
// Per-step fused: softmax + coverage + context + GRU cell + output projection
// One block per batch element.
// ============================================================================
__global__ void __launch_bounds__(256)
step_kernel(int t, const float* __restrict__ scores, const float* __restrict__ feats,
            float* __restrict__ cov, float* __restrict__ hstate,
            const float* __restrict__ emb, const int* __restrict__ target,
            const float* __restrict__ count, const float* __restrict__ wih,
            const float* __restrict__ whh, const float* __restrict__ bih,
            const float* __restrict__ bhh, const float* __restrict__ out_w,
            const float* __restrict__ out_b, float* __restrict__ out) {
  __shared__ float salpha[LDIM];
  __shared__ float red[256];
  __shared__ float xv[XDIM + 1];
  __shared__ float hprev[HDIM];
  int b = blockIdx.x, tid = threadIdx.x;

  // ---- softmax over L=1024 ----
  float sloc[4];
  float mx = -1e30f;
#pragma unroll
  for (int i = 0; i < 4; ++i) {
    sloc[i] = scores[b * LDIM + tid + 256 * i];
    mx = fmaxf(mx, sloc[i]);
  }
  red[tid] = mx; __syncthreads();
  for (int s = 128; s; s >>= 1) { if (tid < s) red[tid] = fmaxf(red[tid], red[tid + s]); __syncthreads(); }
  mx = red[0]; __syncthreads();
  float sum = 0.f;
#pragma unroll
  for (int i = 0; i < 4; ++i) {
    float e = expf(sloc[i] - mx);
    salpha[tid + 256 * i] = e;
    sum += e;
  }
  red[tid] = sum; __syncthreads();
  for (int s = 128; s; s >>= 1) { if (tid < s) red[tid] += red[tid + s]; __syncthreads(); }
  float rinv = 1.f / red[0];
#pragma unroll
  for (int i = 0; i < 4; ++i) {
    int l = tid + 256 * i;
    float a = salpha[l] * rinv;
    salpha[l] = a;
    cov[b * LDIM + l] += a;          // coverage update
  }
  hprev[tid] = hstate[b * HDIM + tid];
  __syncthreads();

  // ---- context[h] = sum_l alpha[l] * feats256[b, l, h] ----
  float ctx = 0.f;
  const float* fb = feats + (size_t)b * LDIM * HDIM + tid;
#pragma unroll 4
  for (int l = 0; l < LDIM; ++l) ctx += salpha[l] * fb[(size_t)l * HDIM];

  // ---- build x = [emb[y], ctx, count] ----
  int y = (t == 0) ? 1 : target[b * TSTEPS + t - 1];
  xv[tid] = emb[y * HDIM + tid];
  xv[HDIM + tid] = ctx;
  if (tid < NCLS) xv[2 * HDIM + tid] = count[b * NCLS + tid];
  __syncthreads();

  // ---- GRU cell: thread tid owns gate index tid ----
  float gr = bih[tid], gz = bih[HDIM + tid], gn = bih[2 * HDIM + tid];
  const float* wr = wih + (size_t)tid * XDIM;
  const float* wz = wih + (size_t)(HDIM + tid) * XDIM;
  const float* wn = wih + (size_t)(2 * HDIM + tid) * XDIM;
#pragma unroll 4
  for (int j = 0; j < XDIM; ++j) {
    float xj = xv[j];
    gr += wr[j] * xj; gz += wz[j] * xj; gn += wn[j] * xj;
  }
  float hr = bhh[tid], hz = bhh[HDIM + tid], hn = bhh[2 * HDIM + tid];
  const float* vr = whh + (size_t)tid * HDIM;
  const float* vz = whh + (size_t)(HDIM + tid) * HDIM;
  const float* vn = whh + (size_t)(2 * HDIM + tid) * HDIM;
#pragma unroll 4
  for (int j = 0; j < HDIM; ++j) {
    float hj = hprev[j];
    hr += vr[j] * hj; hz += vz[j] * hj; hn += vn[j] * hj;
  }
  float r    = 1.f / (1.f + expf(-(gr + hr)));
  float z    = 1.f / (1.f + expf(-(gz + hz)));
  float n    = tanhf(gn + r * hn);
  float hnew = (1.f - z) * n + z * hprev[tid];
  hstate[b * HDIM + tid] = hnew;
  __syncthreads();
  xv[tid] = hnew;                    // x2 = [h_new, ctx, count]
  __syncthreads();

  // ---- output projection (111 x 623) ----
  if (tid < NCLS) {
    float o = out_b[tid];
    const float* ow = out_w + (size_t)tid * XDIM;
#pragma unroll 4
    for (int j = 0; j < XDIM; ++j) o += ow[j] * xv[j];
    out[((size_t)b * TSTEPS + t) * NCLS + tid] = o;
  }
}

// ============================================================================
extern "C" void kernel_launch(void* const* d_in, const int* in_sizes, int n_in,
                              void* d_out, int out_size, void* d_ws, size_t ws_size,
                              hipStream_t stream) {
  const float* fm      = (const float*)d_in[0];
  const float* cnt     = (const float*)d_in[1];
  const int*   tgt     = (const int*)d_in[2];
  const float* conv_w  = (const float*)d_in[3];
  const float* conv_b  = (const float*)d_in[4];
  const float* emb     = (const float*)d_in[5];
  const float* wih     = (const float*)d_in[6];
  const float* whh     = (const float*)d_in[7];
  const float* bih     = (const float*)d_in[8];
  const float* bhh     = (const float*)d_in[9];
  const float* attn_w  = (const float*)d_in[10];
  const float* attn_b  = (const float*)d_in[11];
  const float* attn_v  = (const float*)d_in[12];
  const float* attn_vb = (const float*)d_in[13];
  const float* cov_w   = (const float*)d_in[14];
  const float* cov_b   = (const float*)d_in[15];
  const float* out_w   = (const float*)d_in[16];
  const float* out_b   = (const float*)d_in[17];

  char* ws = (char*)d_ws;
  float* peB    = (float*)ws; ws += (size_t)LDIM * 512 * 4;          // 2 MB
  float* peW    = (float*)ws; ws += (size_t)LDIM * HDIM * 4;         // 1 MB
  float* Wca    = (float*)ws; ws += (size_t)HDIM * CIN * 4;          // 1 MB
  float* feats  = (float*)ws; ws += (size_t)BATCH * LDIM * HDIM * 4; // 64 MB
  float* Wf     = (float*)ws; ws += (size_t)BATCH * LDIM * HDIM * 4; // 64 MB
  float* hstate = (float*)ws; ws += (size_t)BATCH * HDIM * 4;
  float* cov    = (float*)ws; ws += (size_t)BATCH * LDIM * 4;
  float* scores = (float*)ws;

  // precompute (bias tables, fused weight product, zero state)
  peB_kernel<<<LDIM, 256, 0, stream>>>(conv_b, peB);
  wca_kernel<<<dim3(HDIM, CIN / 256), 256, 0, stream>>>(attn_w, conv_w, Wca);
  peW_kernel<<<LDIM, 256, 0, stream>>>(peB, attn_w, attn_b, cov_b, peW);
  init_kernel<<<(BATCH * LDIM) / 256, 256, 0, stream>>>(hstate, cov);

  // WMMA GEMMs: feats256 = fm^T @ conv_w[:256]^T + peB[:, :256]
  //             Wf'      = fm^T @ Wca^T          + peW
  gemm_bf16_wmma<<<dim3(LDIM / 16, BATCH), 256, 0, stream>>>(fm, conv_w, peB, 512, feats);
  gemm_bf16_wmma<<<dim3(LDIM / 16, BATCH), 256, 0, stream>>>(fm, Wca, peW, HDIM, Wf);

  // sequential decode, L2-resident Wf/feats
  for (int t = 0; t < TSTEPS; ++t) {
    scores_kernel<<<(BATCH * LDIM) / 8, 256, 0, stream>>>(Wf, hstate, cov, cov_w,
                                                          attn_v, attn_vb, scores);
    step_kernel<<<BATCH, 256, 0, stream>>>(t, scores, feats, cov, hstate, emb, tgt,
                                           cnt, wih, whh, bih, bhh, out_w, out_b,
                                           (float*)d_out);
  }
}
